// ContrastiveLoss_43130061586994
// MI455X (gfx1250) — 16-bit and 8-bit matrix operand order repaired and checked by reading, compiled, not run on hardware
//
#include <hip/hip_runtime.h>
#include <hip/hip_fp16.h>


#define NN 4096
#define DD 128

typedef __attribute__((ext_vector_type(16))) _Float16 v16h;
typedef __attribute__((ext_vector_type(8)))  _Float16 v8h;
typedef __attribute__((ext_vector_type(8)))  float    v8f;
typedef __attribute__((ext_vector_type(8)))  int      v8i;

__device__ __forceinline__ void async_copy16(unsigned lds_off, const void* gptr) {
  asm volatile("global_load_async_to_lds_b128 %0, %1, off"
               :: "v"(lds_off), "v"(gptr)
               : "memory");
}

__device__ __forceinline__ float blockReduceSum(float v) {
  __shared__ float sm[8];
  __syncthreads();
  #pragma unroll
  for (int o = 16; o > 0; o >>= 1) v += __shfl_xor(v, o, 32);
  int lane = threadIdx.x & 31, w = threadIdx.x >> 5;
  if (lane == 0) sm[w] = v;
  __syncthreads();
  int nw = blockDim.x >> 5;
  v = (threadIdx.x < nw) ? sm[threadIdx.x] : 0.0f;
  if (w == 0)
    #pragma unroll
    for (int o = 4; o > 0; o >>= 1) v += __shfl_xor(v, o, 32);
  return v;
}

__global__ void prep_kernel(const float* __restrict__ z, const float* __restrict__ ze,
                            _Float16* __restrict__ zh, _Float16* __restrict__ zeh,
                            float* __restrict__ sqz, float* __restrict__ sqe) {
  int row = blockIdx.x, lane = threadIdx.x;
  float4 a = ((const float4*)(z  + (size_t)row * DD))[lane];
  float4 b = ((const float4*)(ze + (size_t)row * DD))[lane];
  _Float16* zr = zh  + (size_t)row * DD + lane * 4;
  zr[0] = (_Float16)a.x; zr[1] = (_Float16)a.y; zr[2] = (_Float16)a.z; zr[3] = (_Float16)a.w;
  _Float16* er = zeh + (size_t)row * DD + lane * 4;
  er[0] = (_Float16)b.x; er[1] = (_Float16)b.y; er[2] = (_Float16)b.z; er[3] = (_Float16)b.w;
  float sa = a.x*a.x + a.y*a.y + a.z*a.z + a.w*a.w;
  float sb = b.x*b.x + b.y*b.y + b.z*b.z + b.w*b.w;
  #pragma unroll
  for (int o = 16; o > 0; o >>= 1) { sa += __shfl_xor(sa, o, 32); sb += __shfl_xor(sb, o, 32); }
  if (lane == 0) { sqz[row] = sa; sqe[row] = sb; }
}

__global__ void __launch_bounds__(256)
gram_dist_kernel(const _Float16* __restrict__ X, const float* __restrict__ sq,
                 float* __restrict__ out, int mode) {
  int lane = threadIdx.x & 31;
  int wid  = blockIdx.x * (blockDim.x >> 5) + (threadIdx.x >> 5);
  int ti = wid >> 8;
  int tj = wid & 255;
  int m = lane & 15, h = lane >> 4;
  const _Float16* arow = X + (size_t)(ti * 16 + m) * DD;
  const _Float16* brow = X + (size_t)(tj * 16 + m) * DD;
  v8f c = {};
  #pragma unroll
  for (int kk = 0; kk < DD; kk += 32) {
    v16h a, b;
    a = *(const v16h*)(arow + kk + h * 16);
    b = *(const v16h*)(brow + kk + h * 16);
    c = __builtin_amdgcn_wmma_f32_16x16x32_f16(false, a, false, b, (short)0, c, false, false);
  }
  int j = tj * 16 + m;
  float sqj = sq[j];
  #pragma unroll
  for (int v = 0; v < 8; ++v) {
    int i = ti * 16 + v + 8 * h;
    float d2 = sq[i] + sqj - 2.0f * c[v];
    float d = sqrtf(fmaxf(d2, 0.0f) + 1e-12f);
    if (mode) d = sqrtf(d + 1e-9f);
    out[(size_t)i * NN + j] = d;
  }
}

__global__ void kth_kernel(const float* __restrict__ dmat, float* __restrict__ kth) {
  __shared__ float sm[32 * 5];
  int row = blockIdx.x, lane = threadIdx.x;
  float best[5] = {3.4e38f, 3.4e38f, 3.4e38f, 3.4e38f, 3.4e38f};
  for (int j = lane; j < NN; j += 32) {
    float v = dmat[(size_t)row * NN + j];
    if (v < best[4]) {
      best[4] = v;
      #pragma unroll
      for (int t = 4; t > 0; --t)
        if (best[t] < best[t - 1]) { float tm = best[t]; best[t] = best[t - 1]; best[t - 1] = tm; }
    }
  }
  #pragma unroll
  for (int t = 0; t < 5; ++t) sm[lane * 5 + t] = best[t];
  __syncthreads();
  if (lane == 0) {
    float b2[5] = {3.4e38f, 3.4e38f, 3.4e38f, 3.4e38f, 3.4e38f};
    for (int idx = 0; idx < 160; ++idx) {
      float v = sm[idx];
      if (v < b2[4]) {
        b2[4] = v;
        #pragma unroll
        for (int t = 4; t > 0; --t)
          if (b2[t] < b2[t - 1]) { float tm = b2[t]; b2[t] = b2[t - 1]; b2[t - 1] = tm; }
      }
    }
    kth[row] = b2[4];
  }
}

__global__ void mask_kernel(const float* __restrict__ dmat, const float* __restrict__ kth,
                            unsigned char* __restrict__ Mm, unsigned char* __restrict__ Rm,
                            float* __restrict__ srow, float* __restrict__ rrow) {
  int i = blockIdx.x;
  float kti = kth[i];
  float s = 0.0f, r = 0.0f;
  for (int j = threadIdx.x; j < NN; j += blockDim.x) {
    float dij = dmat[(size_t)i * NN + j];
    float dji = dmat[(size_t)j * NN + i];
    unsigned char mij = (dij <= kth[j]) ? 1 : 0;
    unsigned char mji = (dji <= kti) ? 1 : 0;
    unsigned char rr = mij & mji;
    Mm[(size_t)i * NN + j] = mij;
    Rm[(size_t)i * NN + j] = rr;
    s += (float)mij; r += (float)rr;
  }
  s = blockReduceSum(s);
  r = blockReduceSum(r);
  if (threadIdx.x == 0) { srow[i] = s; rrow[i] = r; }
}

__global__ void __launch_bounds__(256)
iu8_gemm_kernel(const unsigned char* __restrict__ A, const unsigned char* __restrict__ B,
                unsigned char* __restrict__ out8, int* __restrict__ out32, int mode) {
  __shared__ unsigned char Ab[2][128 * 64];
  __shared__ unsigned char Bb[2][128 * 64];
  int t = threadIdx.x;
  int lane = t & 31, w = t >> 5;
  int sti = blockIdx.x >> 5, stj = blockIdx.x & 31;
  int m = lane & 15, h = lane >> 4;
  int crow   = t >> 2;
  int cchunk = (t & 3) * 16;
  const unsigned char* Abase = A + (size_t)(sti * 128) * NN;
  const unsigned char* Bbase = B + (size_t)(stj * 128) * NN;

  v8i zero = {};
  v8i acc[8];
  #pragma unroll
  for (int n = 0; n < 8; ++n) acc[n] = zero;

  auto load_panel = [&](int kb, int buf) {
    #pragma unroll
    for (int it = 0; it < 2; ++it) {
      int row = crow + it * 64;
      unsigned aoff = (unsigned)(size_t)&Ab[buf][row * 64 + cchunk];
      unsigned boff = (unsigned)(size_t)&Bb[buf][row * 64 + cchunk];
      async_copy16(aoff, Abase + (size_t)row * NN + kb + cchunk);
      async_copy16(boff, Bbase + (size_t)row * NN + kb + cchunk);
    }
  };

  load_panel(0, 0);
  for (int kb = 0; kb < NN; kb += 64) {
    int p = (kb >> 6) & 1;
    if (kb + 64 < NN) {
      load_panel(kb + 64, p ^ 1);
      asm volatile("s_wait_asynccnt 4" ::: "memory");
    } else {
      asm volatile("s_wait_asynccnt 0" ::: "memory");
    }
    __syncthreads();

    const unsigned* pa = (const unsigned*)&Ab[p][(w * 16 + m) * 64 + h * 8];
    v8i a;
    a[0] = (int)pa[0];  a[1] = (int)pa[1];
    a[2] = (int)pa[4];  a[3] = (int)pa[5];
    a[4] = (int)pa[8];  a[5] = (int)pa[9];
    a[6] = (int)pa[12]; a[7] = (int)pa[13];
    #pragma unroll
    for (int nt = 0; nt < 8; ++nt) {
      const unsigned* pb = (const unsigned*)&Bb[p][(nt * 16 + m) * 64 + h * 8];
      v8i b;
      b[0] = (int)pb[0]; b[1] = (int)pb[1]; b[2]  = (int)pb[4];  b[3] = (int)pb[5];
      b[4] = (int)pb[8]; b[5] = (int)pb[9]; b[6]  = (int)pb[12]; b[7] = (int)pb[13];
      acc[nt] = __builtin_amdgcn_wmma_i32_16x16x64_iu8(false, a, false, b, acc[nt], false, false);
    }
    __syncthreads();
  }

  #pragma unroll
  for (int nt = 0; nt < 8; ++nt) {
    int j = stj * 128 + nt * 16 + m;
    #pragma unroll
    for (int v = 0; v < 8; ++v) {
      size_t idx = (size_t)(sti * 128 + w * 16 + v + 8 * h) * NN + j;
      if (mode == 0) out8[idx] = (unsigned char)acc[nt][v];
      else           out32[idx] = acc[nt][v];
    }
  }
}

__global__ void rowmean_kernel(const float* __restrict__ dist, float* __restrict__ dmean) {
  int i = blockIdx.x;
  float s = 0.0f;
  for (int j = threadIdx.x; j < NN; j += blockDim.x) s += dist[(size_t)i * NN + j];
  s = blockReduceSum(s);
  if (threadIdx.x == 0) dmean[i] = s * (1.0f / NN);
}

__global__ void rowloss_kernel(const float* __restrict__ dema, const int* __restrict__ C2,
                               const float* __restrict__ dist, const float* __restrict__ srow,
                               const float* __restrict__ rrow, const float* __restrict__ dmean,
                               float* __restrict__ rowloss) {
  int i = blockIdx.x;
  float inv_i = 1.0f / (srow[i] * rrow[i]);
  float dm = dmean[i];
  float acc = 0.0f;
  for (int j = threadIdx.x; j < NN; j += blockDim.x) {
    float p    = __expf(-dema[(size_t)i * NN + j]);
    float s_ij = (float)C2[(size_t)i * NN + j] * inv_i;
    float s_ji = (float)C2[(size_t)j * NN + i] / (srow[j] * rrow[j]);
    float w    = 0.5f * p + 0.25f * (s_ij + s_ji);
    float delta = dist[(size_t)i * NN + j] / dm;
    float mg   = fmaxf(1.0f - delta, 0.0f);
    acc += w * delta * delta + (1.0f - w) * mg * mg;
  }
  acc = blockReduceSum(acc);
  if (threadIdx.x == 0) rowloss[i] = acc * (1.0f / NN);
}

__global__ void sum_kernel(const float* __restrict__ rowloss, float* __restrict__ out) {
  float acc = 0.0f;
  for (int i = threadIdx.x; i < NN; i += blockDim.x) acc += rowloss[i];
  acc = blockReduceSum(acc);
  if (threadIdx.x == 0) out[0] = acc;
}

extern "C" void kernel_launch(void* const* d_in, const int* in_sizes, int n_in,
                              void* d_out, int out_size, void* d_ws, size_t ws_size,
                              hipStream_t stream) {
  const float* z  = (const float*)d_in[0];
  const float* ze = (const float*)d_in[1];
  float* out = (float*)d_out;
  char* ws = (char*)d_ws;

  _Float16* zh   = (_Float16*)(ws + 0);
  _Float16* zeh  = (_Float16*)(ws + (1ull << 20));
  float* sqz     = (float*)(ws + (2ull << 20));
  float* sqe     = (float*)(ws + (2ull << 20) + 1 * 16384);
  float* kth     = (float*)(ws + (2ull << 20) + 2 * 16384);
  float* srow    = (float*)(ws + (2ull << 20) + 3 * 16384);
  float* rrow    = (float*)(ws + (2ull << 20) + 4 * 16384);
  float* dmean   = (float*)(ws + (2ull << 20) + 5 * 16384);
  float* rowloss = (float*)(ws + (2ull << 20) + 6 * 16384);
  float* dema          = (float*)(ws + (4ull << 20));
  unsigned char* maskm = (unsigned char*)(ws + (68ull << 20));
  unsigned char* Rm    = (unsigned char*)(ws + (84ull << 20));
  unsigned char* C1    = (unsigned char*)(ws + (100ull << 20));
  int* C2              = (int*)(ws + (116ull << 20));
  float* dist          = (float*)(ws + (180ull << 20));

  const int GRAM_BLOCKS = (256 * 256) / 8;
  const int IU8_BLOCKS  = 32 * 32;

  prep_kernel<<<NN, 32, 0, stream>>>(z, ze, zh, zeh, sqz, sqe);
  gram_dist_kernel<<<GRAM_BLOCKS, 256, 0, stream>>>(zeh, sqe, dema, 0);
  kth_kernel<<<NN, 32, 0, stream>>>(dema, kth);
  mask_kernel<<<NN, 256, 0, stream>>>(dema, kth, maskm, Rm, srow, rrow);
  iu8_gemm_kernel<<<IU8_BLOCKS, 256, 0, stream>>>(maskm, maskm, C1, nullptr, 0);
  iu8_gemm_kernel<<<IU8_BLOCKS, 256, 0, stream>>>(C1, Rm, nullptr, C2, 1);
  gram_dist_kernel<<<GRAM_BLOCKS, 256, 0, stream>>>(zh, sqz, dist, 1);
  rowmean_kernel<<<NN, 256, 0, stream>>>(dist, dmean);
  rowloss_kernel<<<NN, 256, 0, stream>>>(dema, C2, dist, srow, rrow, dmean, rowloss);
  sum_kernel<<<1, 256, 0, stream>>>(rowloss, out);
}
